// GRUOptimized_33921651704516
// MI455X (gfx1250) — compile-verified, matrix operand order checked by reading, not run on hardware
//
#include <hip/hip_runtime.h>
#include <hip/hip_bf16.h>
#include <math.h>

// ---------------------------------------------------------------------------
// GRU, 6 layers, SEQ=512, B=64, IN=128, H=256  (fp32 in/out, bf16 WMMA inside)
// Phase 1: xg GEMM with TDM-staged A tiles (double buffered) + WMMA bf16
// Phase 2: single-WGP recurrent scan, h in LDS, xg slices prefetched by TDM
// ---------------------------------------------------------------------------
#define SEQ 512
#define BATCH 64
#define INDIM 128
#define HDIM 256
#define G3H 768              // 3*H
#define NLAYERS 6
#define MROWS (SEQ * BATCH)  // 32768
#define XGSLICE (BATCH * G3H)  // 49152 elements per timestep

typedef __attribute__((ext_vector_type(16))) __bf16 v16bf;
typedef __attribute__((ext_vector_type(8)))  __bf16 v8bf;
typedef __attribute__((ext_vector_type(8)))  float  v8f;
typedef __attribute__((ext_vector_type(4)))  unsigned int v4u;
typedef __attribute__((ext_vector_type(8)))  int v8i;
typedef __attribute__((ext_vector_type(4)))  int v4i;

// ---------------------------------------------------------------------------
// Tensor Data Mover: async 2D bf16 tile copy global -> LDS.
// D# bitfields per CDNA5 ISA 8.3/8.4 (group0: count/lds/global/type,
// group1: data_size, tensor dims, tile dims, strides).  Issue once per wave
// (EXEC ignored); completion via TENSORcnt.
// ---------------------------------------------------------------------------
__device__ __forceinline__ void tdm_load_2d_bf16(
    unsigned int lds_off, const void* gptr,
    unsigned int tile_d0, unsigned int tile_d1,       // elements, rows
    unsigned int tensor_d0, unsigned int tensor_d1,   // OOB extents
    unsigned long long d0_stride_elems) {
  unsigned long long ga = (unsigned long long)gptr;
  v4u g0;
  g0[0] = 1u;                                   // count=1 valid descriptor
  g0[1] = lds_off;                              // lds_addr (bytes)
  g0[2] = (unsigned int)ga;                     // global_addr[31:0]
  g0[3] = ((unsigned int)(ga >> 32) & 0x1FFFFFFu) | (2u << 30);  // type=2
  v8i g1;
  g1[0] = (int)(1u << 16);                      // data_size=1 -> 2 bytes
  g1[1] = (int)((tensor_d0 & 0xFFFFu) << 16);   // tensor_dim0[15:0]
  g1[2] = (int)((tensor_d0 >> 16) | ((tensor_d1 & 0xFFFFu) << 16));
  g1[3] = (int)((tensor_d1 >> 16) | ((tile_d0 & 0xFFFFu) << 16));
  g1[4] = (int)(tile_d1 & 0xFFFFu);             // tile_dim1 (tile_dim2=0)
  g1[5] = (int)(unsigned int)(d0_stride_elems & 0xFFFFFFFFull);
  g1[6] = (int)((unsigned int)(d0_stride_elems >> 32) & 0xFFFFu);
  g1[7] = 0;
  v4i z4 = {0, 0, 0, 0};
#if defined(__clang_major__) && __clang_major__ >= 23
  v8i z8 = {0, 0, 0, 0, 0, 0, 0, 0};
  __builtin_amdgcn_tensor_load_to_lds(g0, g1, z4, z4, z8, 0);
#else
  __builtin_amdgcn_tensor_load_to_lds(g0, g1, z4, z4, 0);
#endif
}

__device__ __forceinline__ unsigned int lds_offset_of(const void* p) {
  return (unsigned int)(unsigned long long)(uintptr_t)p;
}

// Load one 16x32 bf16 WMMA fragment.  Per CDNA5 ISA layout:
//   lane<16 : row=lane,    elems[0..7]=K[0..7],  elems[8..15]=K[16..23]
//   lane>=16: row=lane-16, elems[0..7]=K[8..15], elems[8..15]=K[24..31]
__device__ __forceinline__ v16bf ldfrag(const __bf16* __restrict__ base,
                                        int rowStride, int lane) {
  int r  = lane & 15;
  int kh = (lane >> 4) << 3;
  const __bf16* p = base + (size_t)r * rowStride + kh;
  union { v16bf v; v8bf h[2]; } u;
  u.h[0] = *(const v8bf*)(p);
  u.h[1] = *(const v8bf*)(p + 16);
  return u.v;
}

__device__ __forceinline__ v8f wmma_bf16(v16bf a, v16bf b, v8f c) {
  return __builtin_amdgcn_wmma_f32_16x16x32_bf16(
      false, a, false, b, (short)0, c, false, false);
}

// ---------------------------------------------------------------------------
// f32 -> bf16 conversion
// ---------------------------------------------------------------------------
__global__ void cvt_f32_bf16(const float* __restrict__ in,
                             __bf16* __restrict__ out, int n) {
  int i = blockIdx.x * blockDim.x + threadIdx.x;
  if (i < n) out[i] = (__bf16)in[i];
}

// ---------------------------------------------------------------------------
// Phase 1: xg[M,768] = Xbf[M,K] * Wbf[768,K]^T + b_ih  (bf16 WMMA, f32 acc,
// bf16 result).  Block tile 128(M) x 64(N); 8 waves, each a 32x32 sub-tile.
// A tiles DMA'd into LDS by the TDM, double buffered (issue kc+1, compute kc).
// ---------------------------------------------------------------------------
__global__ __launch_bounds__(256) void gemm_xg_kernel(
    const __bf16* __restrict__ X,     // [M,K]
    const __bf16* __restrict__ W,     // [768,K] (row = output col)
    const float*  __restrict__ bias,  // [768]
    __bf16* __restrict__ xg,          // [M,768] bf16
    int K) {
  __shared__ __bf16 As[2][128][32];

  const int tid  = threadIdx.x;
  const int lane = tid & 31;
  const int w    = tid >> 5;
  const int wm   = w & 3;   // 4 m sub-tiles of 32
  const int wn   = w >> 2;  // 2 n sub-tiles of 32
  const int tileM = blockIdx.x * 128;
  const int tileN = blockIdx.y * 64;
  const int n0 = tileN + wn * 32;

  v8f c00 = {}, c01 = {}, c10 = {}, c11 = {};

  const unsigned int ldsA0 = lds_offset_of(&As[0][0][0]);
  const unsigned int ldsA1 = lds_offset_of(&As[1][0][0]);
  const int nkc = K >> 5;

  // Prologue: DMA first A tile (128 rows x 32 cols) into buffer 0.
  if (w == 0) {
    tdm_load_2d_bf16(ldsA0, X + (size_t)tileM * K, 32, 128,
                     (unsigned int)K, 1u << 20, (unsigned long long)K);
  }

  for (int kc = 0; kc < nkc; ++kc) {
    const int k0 = kc << 5;
    if (w == 0) {
      if (kc + 1 < nkc) {
        tdm_load_2d_bf16((kc & 1) ? ldsA0 : ldsA1,
                         X + (size_t)tileM * K + ((kc + 1) << 5), 32, 128,
                         (unsigned int)K, 1u << 20, (unsigned long long)K);
        __builtin_amdgcn_s_wait_tensorcnt((short)1);  // buffer kc ready
      } else {
        __builtin_amdgcn_s_wait_tensorcnt((short)0);
      }
    }
    __syncthreads();  // publish DMA'd tile to all waves

    const __bf16* abase = &As[kc & 1][0][0];
    v16bf a0 = ldfrag(abase + (size_t)(wm * 32) * 32,      32, lane);
    v16bf a1 = ldfrag(abase + (size_t)(wm * 32 + 16) * 32, 32, lane);
    v16bf b0 = ldfrag(W + (size_t)(n0)      * K + k0, K, lane);
    v16bf b1 = ldfrag(W + (size_t)(n0 + 16) * K + k0, K, lane);

    c00 = wmma_bf16(a0, b0, c00);
    c01 = wmma_bf16(a0, b1, c01);
    c10 = wmma_bf16(a1, b0, c10);
    c11 = wmma_bf16(a1, b1, c11);
    __syncthreads();  // all reads done before buffer is DMA-overwritten
  }

  // Epilogue: C/D layout -> lane l, reg r : row=(l>>4)*8+r, col=l&15
  const int cr = (lane >> 4) << 3;
  const int cc = lane & 15;
  const int m0 = tileM + wm * 32;
#pragma unroll
  for (int sm = 0; sm < 2; ++sm) {
#pragma unroll
    for (int sn = 0; sn < 2; ++sn) {
      v8f acc = sm == 0 ? (sn == 0 ? c00 : c01) : (sn == 0 ? c10 : c11);
      const int gn = n0 + sn * 16 + cc;
      const float b = bias[gn];
#pragma unroll
      for (int r = 0; r < 8; ++r) {
        const int gm = m0 + sm * 16 + cr + r;
        xg[(size_t)gm * G3H + gn] = (__bf16)(acc[r] + b);
      }
    }
  }
}

// ---------------------------------------------------------------------------
// Phase 2: recurrent scan, ONE workgroup (32 wave32 on one WGP).
// h lives in LDS (f32 master + bf16 WMMA copy).  The xg slice for step t+1
// is TDM-prefetched into LDS while step t's WMMAs run -> HBM latency off the
// serial critical path.  LDS: 64K hF + 32K hB + 2x96K xg = 288K (<=320K/WGP).
// ---------------------------------------------------------------------------
__global__ __launch_bounds__(1024) void gru_scan_kernel(
    const __bf16* __restrict__ xg,    // [512*64, 768] bf16, b_ih added
    const __bf16* __restrict__ Whh,   // [768, 256] bf16
    const float*  __restrict__ bhh,   // [768]
    const float*  __restrict__ h0,    // [64, 256]
    __bf16* __restrict__ y_bf,        // [512*64,256] bf16 (next layer) or null
    float*  __restrict__ y_f32,       // [512*64,256] f32 (final layer) or null
    float*  __restrict__ h_out) {     // [64,256] final hidden
  __shared__ float  hF[BATCH * HDIM];       // 64 KB f32 master copy of h
  __shared__ __bf16 hB[BATCH * HDIM];       // 32 KB bf16 copy for WMMA A
  __shared__ __bf16 xgS[2][XGSLICE];        // 2 x 96 KB prefetched xg slices

  const int tid  = threadIdx.x;
  const int lane = tid & 31;
  const int w    = tid >> 5;

  for (int i = tid; i < BATCH * HDIM; i += 1024) {
    float v = h0[i];
    hF[i] = v;
    hB[i] = (__bf16)v;
  }
  // Prefetch xg slice for t=0 (contiguous 49152 bf16 = one 1D TDM tile).
  if (w == 0) {
    tdm_load_2d_bf16(lds_offset_of(&xgS[0][0]), xg, XGSLICE, 0,
                     XGSLICE, 0, (unsigned long long)XGSLICE);
    __builtin_amdgcn_s_wait_tensorcnt((short)0);
  }
  __syncthreads();

  const int cr = (lane >> 4) << 3;  // acc row offset within 16
  const int cc = lane & 15;         // acc col within 16

  // Per-unit constants (unit u: m = u&3, triple = u>>2)
  float bR[2], bZ[2], bN[2];
#pragma unroll
  for (int ui = 0; ui < 2; ++ui) {
    const int u = w * 2 + ui;
    const int h = (u >> 2) * 16 + cc;
    bR[ui] = bhh[h];
    bZ[ui] = bhh[HDIM + h];
    bN[ui] = bhh[2 * HDIM + h];
  }

  for (int t = 0; t < SEQ; ++t) {
    // Kick off DMA of the NEXT xg slice; it completes under this step's WMMAs.
    if (w == 0 && t + 1 < SEQ) {
      tdm_load_2d_bf16(lds_offset_of(&xgS[(t + 1) & 1][0]),
                       xg + (size_t)(t + 1) * XGSLICE, XGSLICE, 0,
                       XGSLICE, 0, (unsigned long long)XGSLICE);
    }

    float hnew[2][8];
#pragma unroll
    for (int ui = 0; ui < 2; ++ui) {
      const int u    = w * 2 + ui;
      const int m    = u & 3;    // batch tile (16 rows)
      const int trip = u >> 2;   // hidden-col tile (16 cols)
      v8f ar = {}, az = {}, an = {};
      const __bf16* hbase = &hB[(size_t)(m * 16) * HDIM];
      const __bf16* wR = Whh + (size_t)(trip * 16) * HDIM;
      const __bf16* wZ = Whh + (size_t)(HDIM + trip * 16) * HDIM;
      const __bf16* wN = Whh + (size_t)(2 * HDIM + trip * 16) * HDIM;
#pragma unroll
      for (int kc = 0; kc < 8; ++kc) {
        const int k0 = kc << 5;
        v16bf a  = ldfrag(hbase + k0, HDIM, lane);
        v16bf fr = ldfrag(wR + k0, HDIM, lane);
        v16bf fz = ldfrag(wZ + k0, HDIM, lane);
        v16bf fn = ldfrag(wN + k0, HDIM, lane);
        ar = wmma_bf16(a, fr, ar);
        az = wmma_bf16(a, fz, az);
        an = wmma_bf16(a, fn, an);
      }
      // Gates (f32) -- xg comes from the LDS slice prefetched last step.
      const int hcol = trip * 16 + cc;
      const __bf16* xgt = &xgS[t & 1][0];
#pragma unroll
      for (int r = 0; r < 8; ++r) {
        const int b = m * 16 + cr + r;
        const __bf16* xrow = xgt + (size_t)b * G3H;
        const float xr = (float)xrow[hcol];
        const float xz = (float)xrow[HDIM + hcol];
        const float xn = (float)xrow[2 * HDIM + hcol];
        const float hr = ar[r] + bR[ui];
        const float hz = az[r] + bZ[ui];
        const float hn = an[r] + bN[ui];
        const float rg = 1.0f / (1.0f + __expf(-(xr + hr)));
        const float zg = 1.0f / (1.0f + __expf(-(xz + hz)));
        const float ng = tanhf(xn + rg * hn);
        const float ho = hF[b * HDIM + hcol];
        hnew[ui][r] = (1.0f - zg) * ng + zg * ho;
      }
    }
    __syncthreads();  // everyone done reading old h
#pragma unroll
    for (int ui = 0; ui < 2; ++ui) {
      const int u    = w * 2 + ui;
      const int m    = u & 3;
      const int trip = u >> 2;
      const int hcol = trip * 16 + cc;
#pragma unroll
      for (int r = 0; r < 8; ++r) {
        const int b = m * 16 + cr + r;
        const float v = hnew[ui][r];
        hF[b * HDIM + hcol] = v;
        hB[b * HDIM + hcol] = (__bf16)v;
        const size_t yi = ((size_t)t * BATCH + b) * HDIM + hcol;
        if (y_bf)  y_bf[yi]  = (__bf16)v;
        if (y_f32) y_f32[yi] = v;
      }
    }
    // Next step's xg slice must be fully in LDS before anyone reads it.
    if (w == 0 && t + 1 < SEQ) __builtin_amdgcn_s_wait_tensorcnt((short)0);
    __syncthreads();
  }

  for (int i = tid; i < BATCH * HDIM; i += 1024) h_out[i] = hF[i];
}

// ---------------------------------------------------------------------------
// Host side
// ---------------------------------------------------------------------------
extern "C" void kernel_launch(void* const* d_in, const int* in_sizes, int n_in,
                              void* d_out, int out_size, void* d_ws, size_t ws_size,
                              hipStream_t stream) {
  const float* x     = (const float*)d_in[0];  // [512,64,128]
  const float* h0    = (const float*)d_in[1];  // [6,64,256]
  const float* w_ih0 = (const float*)d_in[2];  // [768,128]
  const float* w_ihr = (const float*)d_in[3];  // [5,768,256]
  const float* w_hh  = (const float*)d_in[4];  // [6,768,256]
  const float* b_ih  = (const float*)d_in[5];  // [6,768]
  const float* b_hh  = (const float*)d_in[6];  // [6,768]

  float* out_y  = (float*)d_out;                      // [512,64,256]
  float* out_hn = out_y + (size_t)SEQ * BATCH * HDIM; // [6,64,256]

  // Workspace layout (~82 MB)
  char*   ws   = (char*)d_ws;
  __bf16* xg   = (__bf16*)ws;                          // 32768*768 bf16 (48MB)
  size_t  off  = (size_t)MROWS * G3H * sizeof(__bf16);
  __bf16* actA = (__bf16*)(ws + off); off += (size_t)MROWS * HDIM * 2;
  __bf16* actB = (__bf16*)(ws + off); off += (size_t)MROWS * HDIM * 2;
  __bf16* wihB = (__bf16*)(ws + off); off += (size_t)G3H * HDIM * 2;
  __bf16* whhB = (__bf16*)(ws + off);

  // Layer-0 input -> bf16
  {
    int n = MROWS * INDIM;
    cvt_f32_bf16<<<(n + 255) / 256, 256, 0, stream>>>(x, actA, n);
  }

  for (int l = 0; l < NLAYERS; ++l) {
    const int K = (l == 0) ? INDIM : HDIM;
    const float* wih = (l == 0) ? w_ih0 : (w_ihr + (size_t)(l - 1) * G3H * HDIM);

    int nw = G3H * K;
    cvt_f32_bf16<<<(nw + 255) / 256, 256, 0, stream>>>(wih, wihB, nw);
    int nh = G3H * HDIM;
    cvt_f32_bf16<<<(nh + 255) / 256, 256, 0, stream>>>(
        w_hh + (size_t)l * G3H * HDIM, whhB, nh);

    const __bf16* xin = (l & 1) ? actB : actA;
    __bf16* ybf = (l == NLAYERS - 1) ? nullptr : ((l & 1) ? actA : actB);
    float*  yf  = (l == NLAYERS - 1) ? out_y : nullptr;

    dim3 grid(MROWS / 128, G3H / 64);
    gemm_xg_kernel<<<grid, 256, 0, stream>>>(xin, wihB, b_ih + (size_t)l * G3H,
                                             xg, K);

    gru_scan_kernel<<<1, 1024, 0, stream>>>(
        xg, whhB, b_hh + (size_t)l * G3H, h0 + (size_t)l * BATCH * HDIM,
        ybf, yf, out_hn + (size_t)l * BATCH * HDIM);
  }
}
